// MultiHeadAttention_84782654423450
// MI455X (gfx1250) — matrix operand order repaired and checked by reading, not compiled, not run on hardware
//
#include <hip/hip_runtime.h>
#include <hip/hip_bf16.h>

// ---------------------------------------------------------------------------
// CDNA5 (gfx1250) multi-head attention, bf16 WMMA path, fp32 accumulate.
// All GEMM inner loops are pure b128-load -> v_wmma_f32_16x16x32_bf16;
// fp32->bf16 conversion is hoisted into one streaming pre-pass.
// ---------------------------------------------------------------------------

typedef __attribute__((ext_vector_type(16))) __bf16 v16bf;
typedef __attribute__((ext_vector_type(8)))  float  v8f;

union FragB16 {
    v16bf v;
    unsigned short u[16];
    uint4 q[2];
};

// fp32 -> bf16, round-to-nearest-even.
__device__ __forceinline__ unsigned short f2bf(float f) {
    union { float f; unsigned u; } x;
    x.f = f;
    unsigned r = x.u + 0x7FFFu + ((x.u >> 16) & 1u);
    return (unsigned short)(r >> 16);
}

__device__ __forceinline__ v8f wmma_bf16(const FragB16& a, const FragB16& b, v8f c) {
    return __builtin_amdgcn_wmma_f32_16x16x32_bf16(
        /*neg_a=*/false, a.v, /*neg_b=*/false, b.v,
        /*c_mod=*/(short)0, c, /*reuse_a=*/false, /*reuse_b=*/false);
}

// Problem constants (fixed by the reference).
#define BB   2
#define SS   2048
#define DD   1024
#define HH   16
#define DK   64
#define MTOT (BB * SS)   // 4096

// ---------------------------------------------------------------------------
// Streaming fp32 -> bf16 conversion (8 elements / thread, b128 in+out).
// n must be a multiple of 2048 (true for all tensors here).
// ---------------------------------------------------------------------------
__global__ __launch_bounds__(256) void cvt_bf16_kernel(
    const float* __restrict__ in, unsigned short* __restrict__ out)
{
    const size_t i = ((size_t)blockIdx.x * 256 + threadIdx.x) * 8;
    float4 a = *(const float4*)(in + i);
    float4 b = *(const float4*)(in + i + 4);
    unsigned short tmp[8] = { f2bf(a.x), f2bf(a.y), f2bf(a.z), f2bf(a.w),
                              f2bf(b.x), f2bf(b.y), f2bf(b.z), f2bf(b.w) };
    *(uint4*)(out + i) = *(const uint4*)tmp;
}

// ---------------------------------------------------------------------------
// GEMM: Y[M,N] = X[M,K] @ W[N,K]^T + bias[N]   (X, W bf16; acc fp32)
//   MODE 0: Y bf16 head-split [B,H,S,dk]    (Q, K projections)
//   MODE 1: Y bf16 transposed [B,H,dk,S]    (V projection)
//   MODE 2: Y fp32 plain [M,N]              (output projection)
// Block = 256 threads = 8 waves, block tile 128x128, wave tile 32x64,
// k-step 32 -> 12 b128 loads + 8 WMMAs per wave per k-step.
// ---------------------------------------------------------------------------
template <int MODE>
__global__ __launch_bounds__(256) void gemm_wmma_kernel(
    const unsigned short* __restrict__ X, const unsigned short* __restrict__ W,
    const float* __restrict__ bias, void* __restrict__ Yv)
{
    constexpr int N = DD, K = DD;
    const int lane  = threadIdx.x & 31;
    const int wave  = threadIdx.x >> 5;
    const int hi    = lane >> 4;      // 0/1: lane half
    const int lo    = lane & 15;
    const int mBase = blockIdx.y * 128 + (wave >> 1) * 32;
    const int nBase = blockIdx.x * 128 + (wave & 1) * 64;

    v8f acc[2][4];
#pragma unroll
    for (int im = 0; im < 2; ++im)
#pragma unroll
        for (int in_ = 0; in_ < 4; ++in_) acc[im][in_] = {};

    for (int kk = 0; kk < K; kk += 32) {
        // A fragments: 16x32. Lane = row, 16 contiguous k at kk + 16*hi.
        FragB16 a[2];
#pragma unroll
        for (int im = 0; im < 2; ++im) {
            const unsigned short* p =
                X + (size_t)(mBase + im * 16 + lo) * K + kk + 16 * hi;
            a[im].q[0] = *(const uint4*)(p);
            a[im].q[1] = *(const uint4*)(p + 8);
        }
        // B fragments: 32x16, element (k,n) = W[n,k] -> contiguous slice of
        // row n of W. Lane = col, 16 contiguous k at kk + 16*hi.
        FragB16 bf[4];
#pragma unroll
        for (int in_ = 0; in_ < 4; ++in_) {
            const unsigned short* p =
                W + (size_t)(nBase + in_ * 16 + lo) * K + kk + 16 * hi;
            bf[in_].q[0] = *(const uint4*)(p);
            bf[in_].q[1] = *(const uint4*)(p + 8);
        }
#pragma unroll
        for (int im = 0; im < 2; ++im)
#pragma unroll
            for (int in_ = 0; in_ < 4; ++in_)
                acc[im][in_] = wmma_bf16(a[im], bf[in_], acc[im][in_]);
    }

    // Epilogue. C layout: element (m,n) -> lane lo = n%16, m = v + 8*hi.
#pragma unroll
    for (int im = 0; im < 2; ++im) {
#pragma unroll
        for (int in_ = 0; in_ < 4; ++in_) {
            const int n  = nBase + in_ * 16 + lo;
            const float bv = bias[n];
            if (MODE == 2) {
                float* Y = (float*)Yv;
#pragma unroll
                for (int v = 0; v < 8; ++v) {
                    const int m = mBase + im * 16 + v + 8 * hi;
                    Y[(size_t)m * N + n] = acc[im][in_][v] + bv;
                }
            } else if (MODE == 0) {
                unsigned short* Y = (unsigned short*)Yv;
                const int h = n >> 6, d = n & 63;
#pragma unroll
                for (int v = 0; v < 8; ++v) {
                    const int m = mBase + im * 16 + v + 8 * hi;
                    const int b = m >> 11, s = m & (SS - 1);
                    Y[(((size_t)(b * HH + h) * SS + s) << 6) + d] =
                        f2bf(acc[im][in_][v] + bv);
                }
            } else {  // MODE 1: V transposed [B,H,dk,S]; 8 consecutive s -> b128
                unsigned short* Y = (unsigned short*)Yv;
                const int h  = n >> 6, d = n & 63;
                const int s0 = mBase + im * 16 + 8 * hi;     // 8-aligned
                const int b  = s0 >> 11, s = s0 & (SS - 1);
                unsigned short tmp[8];
#pragma unroll
                for (int v = 0; v < 8; ++v) tmp[v] = f2bf(acc[im][in_][v] + bv);
                *(uint4*)&Y[(((size_t)(b * HH + h) << 6) + d) * SS + s] =
                    *(const uint4*)tmp;
            }
        }
    }
}

// ---------------------------------------------------------------------------
// Flash attention: grid (S/128, H, B); 8 waves/block, 16 query rows per wave.
// KV streamed in steps of 32 columns, online softmax, P routed through LDS
// (C-layout -> A-layout transpose), P@V via WMMA against transposed V.
// ---------------------------------------------------------------------------
__global__ __launch_bounds__(256) void attn_kernel(
    const unsigned short* __restrict__ Q,   // [B,H,S,dk] bf16
    const unsigned short* __restrict__ Kc,  // [B,H,S,dk] bf16
    const unsigned short* __restrict__ VT,  // [B,H,dk,S] bf16
    const int* __restrict__ mask,           // [B,S,S]
    unsigned short* __restrict__ ctx)       // [B,S,D] bf16 (heads merged)
{
    __shared__ __align__(16) unsigned short plds[8][16][32];

    const int lane = threadIdx.x & 31;
    const int wave = threadIdx.x >> 5;
    const int hi = lane >> 4, lo = lane & 15;
    const int b = blockIdx.z, h = blockIdx.y;
    const int q0 = blockIdx.x * 128 + wave * 16;

    const size_t bh = (size_t)(b * HH + h);
    const unsigned short* Qbh = Q  + bh * SS * DK;
    const unsigned short* Kbh = Kc + bh * SS * DK;
    const unsigned short* Vbh = VT + bh * DK * SS;
    const int* Mb = mask + (size_t)b * SS * SS;

    // Resident Q fragments (16 rows x dk=64 = two 16x32 A-frags).
    FragB16 qf[2];
    {
        const unsigned short* qp = Qbh + (size_t)(q0 + lo) * DK;
#pragma unroll
        for (int ks = 0; ks < 2; ++ks) {
            qf[ks].q[0] = *(const uint4*)(qp + ks * 32 + 16 * hi);
            qf[ks].q[1] = *(const uint4*)(qp + ks * 32 + 16 * hi + 8);
        }
    }

    v8f acc[4];
#pragma unroll
    for (int t = 0; t < 4; ++t) acc[t] = {};
    float mrun[8], lrun[8];
#pragma unroll
    for (int v = 0; v < 8; ++v) { mrun[v] = -1e30f; lrun[v] = 0.0f; }

    for (int kv = 0; kv < SS; kv += 32) {
        if (kv + 32 < SS) {  // hint next KV tiles toward the caches
            __builtin_prefetch(Kbh + (size_t)(kv + 32 + lo) * DK, 0, 1);
            __builtin_prefetch(Vbh + (size_t)lo * SS + kv + 32, 0, 1);
        }

        // scores: S(16x32) = Q(16x64) @ K_tile(32x64)^T, two 16x16 tiles.
        v8f s0 = {}, s1 = {};
#pragma unroll
        for (int ks = 0; ks < 2; ++ks) {
            FragB16 k0, k1;
            const unsigned short* kp0 = Kbh + (size_t)(kv + lo) * DK + ks * 32 + 16 * hi;
            k0.q[0] = *(const uint4*)(kp0);
            k0.q[1] = *(const uint4*)(kp0 + 8);
            const unsigned short* kp1 = Kbh + (size_t)(kv + 16 + lo) * DK + ks * 32 + 16 * hi;
            k1.q[0] = *(const uint4*)(kp1);
            k1.q[1] = *(const uint4*)(kp1 + 8);
            s0 = wmma_bf16(qf[ks], k0, s0);
            s1 = wmma_bf16(qf[ks], k1, s1);
        }

        // Online softmax per row (row m = v + 8*hi, spread over 16 lanes).
#pragma unroll
        for (int v = 0; v < 8; ++v) {
            const int qrow = q0 + v + 8 * hi;
            const int* mp = Mb + (size_t)qrow * SS + kv;
            float x0 = s0[v] * 0.125f;           // 1/sqrt(64)
            float x1 = s1[v] * 0.125f;
            if (mp[lo]      == 0) x0 = -1e30f;
            if (mp[16 + lo] == 0) x1 = -1e30f;

            float rmax = fmaxf(x0, x1);
            rmax = fmaxf(rmax, __shfl_xor(rmax, 1, 16));
            rmax = fmaxf(rmax, __shfl_xor(rmax, 2, 16));
            rmax = fmaxf(rmax, __shfl_xor(rmax, 4, 16));
            rmax = fmaxf(rmax, __shfl_xor(rmax, 8, 16));

            const float mnew  = fmaxf(mrun[v], rmax);
            const float alpha = __expf(mrun[v] - mnew);
            const float p0 = __expf(x0 - mnew);
            const float p1 = __expf(x1 - mnew);
            float rs = p0 + p1;
            rs += __shfl_xor(rs, 1, 16);
            rs += __shfl_xor(rs, 2, 16);
            rs += __shfl_xor(rs, 4, 16);
            rs += __shfl_xor(rs, 8, 16);

            lrun[v] = lrun[v] * alpha + rs;
            mrun[v] = mnew;
#pragma unroll
            for (int t = 0; t < 4; ++t) acc[t][v] *= alpha;

            // C-layout -> LDS (row = v + 8*hi, col = lane's column).
            plds[wave][v + 8 * hi][lo]      = f2bf(p0);
            plds[wave][v + 8 * hi][lo + 16] = f2bf(p1);
        }

        // Wave-private LDS: DS ops from one wave are in-order; make sure the
        // stores complete before the transposed reads.
        asm volatile("s_wait_dscnt 0" ::: "memory");

        // P as A-fragment (16x32): lane = row lo, 16 contiguous k at 16*hi.
        FragB16 pf;
        pf.q[0] = *(const uint4*)&plds[wave][lo][16 * hi];
        pf.q[1] = *(const uint4*)&plds[wave][lo][16 * hi + 8];

        // acc(16x64) += P(16x32) @ V_tile(32x64); V is pre-transposed so the
        // B-fragments are contiguous 16-element runs along kv.
#pragma unroll
        for (int t = 0; t < 4; ++t) {
            FragB16 vf;
            const unsigned short* vp = Vbh + (size_t)(t * 16 + lo) * SS + kv + 16 * hi;
            vf.q[0] = *(const uint4*)(vp);
            vf.q[1] = *(const uint4*)(vp + 8);
            acc[t] = wmma_bf16(pf, vf, acc[t]);
        }
    }

    // Normalize and write merged-head context [B,S,D] as bf16.
#pragma unroll
    for (int t = 0; t < 4; ++t) {
#pragma unroll
        for (int v = 0; v < 8; ++v) {
            const int qrow = q0 + v + 8 * hi;
            const float inv = 1.0f / lrun[v];
            ctx[((size_t)(b * SS + qrow)) * DD + h * DK + t * 16 + lo] =
                f2bf(acc[t][v] * inv);
        }
    }
}

// ---------------------------------------------------------------------------
// Host-side launcher.
// Inputs: 0 query, 1 key, 2 value, 3 mask, 4 Wq, 5 bq, 6 Wk, 7 bk,
//         8 Wv, 9 bv, 10 Wo, 11 bo.
// Workspace layout (ushort elements):
//   qb(4M) kb(4M) vtb(4M) xbuf(4M, reused for q/k/v activations, then ctx)
//   wbuf(1M, reused for each weight)                      => 34 MB total.
// ---------------------------------------------------------------------------
extern "C" void kernel_launch(void* const* d_in, const int* in_sizes, int n_in,
                              void* d_out, int out_size, void* d_ws, size_t ws_size,
                              hipStream_t stream) {
    (void)in_sizes; (void)n_in; (void)out_size; (void)ws_size;

    const float* query = (const float*)d_in[0];
    const float* key_  = (const float*)d_in[1];
    const float* value = (const float*)d_in[2];
    const int*   mask  = (const int*)d_in[3];
    const float* Wq = (const float*)d_in[4];
    const float* bq = (const float*)d_in[5];
    const float* Wk = (const float*)d_in[6];
    const float* bk = (const float*)d_in[7];
    const float* Wv = (const float*)d_in[8];
    const float* bv = (const float*)d_in[9];
    const float* Wo = (const float*)d_in[10];
    const float* bo = (const float*)d_in[11];

    const size_t actE = (size_t)MTOT * DD;   // 4M elements
    const size_t wE   = (size_t)DD * DD;     // 1M elements
    unsigned short* qb   = (unsigned short*)d_ws;
    unsigned short* kb   = qb  + actE;
    unsigned short* vtb  = kb  + actE;
    unsigned short* xbuf = vtb + actE;       // activations, later ctx
    unsigned short* wbuf = xbuf + actE;      // current weight (bf16)
    unsigned short* ctx  = xbuf;             // alias: safe, stream-ordered

    dim3 blk(256);
    dim3 gemm_grid(DD / 128, MTOT / 128);            // (8, 32)
    const unsigned cvtA = (unsigned)(actE / (8 * 256));  // 2048 blocks
    const unsigned cvtW = (unsigned)(wE   / (8 * 256));  // 512 blocks

    // Q projection
    cvt_bf16_kernel<<<cvtA, blk, 0, stream>>>(query, xbuf);
    cvt_bf16_kernel<<<cvtW, blk, 0, stream>>>(Wq, wbuf);
    gemm_wmma_kernel<0><<<gemm_grid, blk, 0, stream>>>(xbuf, wbuf, bq, qb);
    // K projection
    cvt_bf16_kernel<<<cvtA, blk, 0, stream>>>(key_, xbuf);
    cvt_bf16_kernel<<<cvtW, blk, 0, stream>>>(Wk, wbuf);
    gemm_wmma_kernel<0><<<gemm_grid, blk, 0, stream>>>(xbuf, wbuf, bk, kb);
    // V projection (transposed output)
    cvt_bf16_kernel<<<cvtA, blk, 0, stream>>>(value, xbuf);
    cvt_bf16_kernel<<<cvtW, blk, 0, stream>>>(Wv, wbuf);
    gemm_wmma_kernel<1><<<gemm_grid, blk, 0, stream>>>(xbuf, wbuf, bv, vtb);

    // Attention (writes ctx == xbuf; all reads of xbuf completed above)
    attn_kernel<<<dim3(SS / 128, HH, BB), blk, 0, stream>>>(qb, kb, vtb, mask, ctx);

    // Output projection -> fp32 d_out
    cvt_bf16_kernel<<<cvtW, blk, 0, stream>>>(Wo, wbuf);
    gemm_wmma_kernel<2><<<gemm_grid, blk, 0, stream>>>(ctx, wbuf, bo, (float*)d_out);
}
